// ImageTransformerDenoiserModel_55190329753684
// MI455X (gfx1250) — matrix operand order repaired and checked by reading, not compiled, not run on hardware
//
#include <hip/hip_runtime.h>
#include <math.h>

// ---------- types ----------
typedef __attribute__((ext_vector_type(16))) __bf16 v16bf;
typedef __attribute__((ext_vector_type(8)))  float  v8f;

struct __attribute__((aligned(16))) U4 { unsigned int x, y, z, w; };
struct __attribute__((aligned(8)))  U2 { unsigned int x, y; };

union FragBF {
    U4 u[2];
    unsigned short s[16];
    v16bf v;
};

// hardware bf16 convert (RNE) via __bf16 cast -> v_cvt_*bf16_f32
__device__ __forceinline__ unsigned short f2bf(float x) {
    union { __bf16 b; unsigned short s; } cv;
    cv.b = (__bf16)x;
    return cv.s;
}

__device__ __forceinline__ v8f zero8() {
    v8f r;
#pragma unroll
    for (int i = 0; i < 8; ++i) r[i] = 0.0f;
    return r;
}

__device__ __forceinline__ v8f wmma_bf16(const FragBF& a, const FragBF& b, v8f c) {
    return __builtin_amdgcn_wmma_f32_16x16x32_bf16(false, a.v, false, b.v, (short)0, c,
                                                   false, false);
}

// A fragment (16x32 bf16): lane holds row M=lane%16; K = half*16 + e, contiguous 16
__device__ __forceinline__ void load_a_bf16(FragBF& a, const unsigned short* row, int kbase) {
    a.u[0] = *(const U4*)(row + kbase);
    a.u[1] = *(const U4*)(row + kbase + 8);
}

// B fragment (32x16 bf16): lane holds col N=lane%16; K = half*16 + e, contiguous 16
__device__ __forceinline__ void load_b_bf16(FragBF& b, const unsigned short* row, int kbase) {
    b.u[0] = *(const U4*)(row + kbase);
    b.u[1] = *(const U4*)(row + kbase + 8);
}

// ---------- K0: one-shot f32 -> bf16 weight conversion ----------
__global__ __launch_bounds__(256) void k0_cvt(const float* __restrict__ src,
                                              unsigned short* __restrict__ dst, int n) {
    int i = (blockIdx.x * 256 + threadIdx.x) * 4;
    if (i >= n) return;
    float4 f = *(const float4*)(src + i);
    unsigned short tmp[4] __attribute__((aligned(8)));
    tmp[0] = f2bf(f.x); tmp[1] = f2bf(f.y); tmp[2] = f2bf(f.z); tmp[3] = f2bf(f.w);
    *(U2*)(dst + i) = *(U2*)tmp;
}

// ---------- K1: c = silu(cond) @ mod_w.T + mod_b  (B=2, 768 cols) ----------
__global__ __launch_bounds__(256) void k1_mod(const float* __restrict__ cond,
                                              const float* __restrict__ mod_w,
                                              const float* __restrict__ mod_b,
                                              float* __restrict__ c_ws) {
    int idx = blockIdx.x * 256 + threadIdx.x;
    if (idx >= 2 * 768) return;
    int b = idx / 768, j = idx % 768;
    float acc = mod_b[j];
    const float* cw = cond + b * 768;
    const float* wr = mod_w + (size_t)j * 768;
    for (int k = 0; k < 768; k += 4) {
        float4 c4 = *(const float4*)(cw + k);
        float4 w4 = *(const float4*)(wr + k);
        float s;
        s = c4.x / (1.f + __expf(-c4.x)); acc += s * w4.x;
        s = c4.y / (1.f + __expf(-c4.y)); acc += s * w4.y;
        s = c4.z / (1.f + __expf(-c4.z)); acc += s * w4.z;
        s = c4.w / (1.f + __expf(-c4.w)); acc += s * w4.w;
    }
    c_ws[idx] = acc;
}

// ---------- K2: xn = rmsnorm(x)*(1+sc)+sh -> bf16; one wave32 per token ----------
__global__ __launch_bounds__(256) void k2_xn(const float* __restrict__ x_in,
                                             const float* __restrict__ norm_scale,
                                             const float* __restrict__ c_ws,
                                             unsigned short* __restrict__ xn) {
    int tid = threadIdx.x;
    int wave = tid >> 5, lane = tid & 31;
    int m = blockIdx.x * 8 + wave;
    int b = m >> 16;
    const float* row = x_in + (size_t)m * 256;
    float4 v0 = ((const float4*)row)[lane * 2];
    float4 v1 = ((const float4*)row)[lane * 2 + 1];
    float ss = v0.x * v0.x + v0.y * v0.y + v0.z * v0.z + v0.w * v0.w +
               v1.x * v1.x + v1.y * v1.y + v1.z * v1.z + v1.w * v1.w;
#pragma unroll
    for (int off = 16; off > 0; off >>= 1) ss += __shfl_xor(ss, off, 32);
    float rs = rsqrtf(ss * (1.0f / 256.0f) + 1e-6f);
    int k0 = lane * 8;
    float xe[8] = {v0.x, v0.y, v0.z, v0.w, v1.x, v1.y, v1.z, v1.w};
    unsigned short tmp[8] __attribute__((aligned(16)));
#pragma unroll
    for (int e = 0; e < 8; ++e) {
        int k = k0 + e;
        float sh = c_ws[b * 768 + k];
        float sc = c_ws[b * 768 + 256 + k];
        float xv = xe[e] * norm_scale[k] * rs;
        tmp[e] = f2bf(xv * (1.f + sc) + sh);
    }
    *(U4*)(xn + (size_t)m * 256 + k0) = *(U4*)tmp;
}

// ---------- K3: QKV GEMM (16 tokens x 768) + QK-norm + RoPE + window scatter ----------
__global__ __launch_bounds__(256) void k3_qkv(const unsigned short* __restrict__ qkvw_bf,
                                              const float* __restrict__ pos,
                                              const float* __restrict__ attn_scale,
                                              const unsigned short* __restrict__ xn,
                                              unsigned short* __restrict__ qn,
                                              unsigned short* __restrict__ kn,
                                              unsigned short* __restrict__ vv) {
    __shared__ float qs[16][776];
    int tid = threadIdx.x;
    int wave = tid >> 5, lane = tid & 31;
    int ln = lane & 15, lh = lane >> 4;
    int m0 = blockIdx.x * 16;

    // ---- GEMM: out[16 x 768] = xn[16 x 256] @ qkv_w.T ----
    v8f acc[6];
#pragma unroll
    for (int i = 0; i < 6; ++i) acc[i] = zero8();
    const unsigned short* arow = xn + (size_t)(m0 + ln) * 256;
    for (int kk = 0; kk < 256; kk += 32) {
        FragBF a;
        load_a_bf16(a, arow, kk + lh * 16);
#pragma unroll
        for (int i = 0; i < 6; ++i) {
            int n = (wave * 6 + i) * 16 + ln;
            FragBF bf;
            load_b_bf16(bf, qkvw_bf + (size_t)n * 256, kk + lh * 16);
            acc[i] = wmma_bf16(a, bf, acc[i]);
        }
    }
#pragma unroll
    for (int i = 0; i < 6; ++i) {
        int n0 = (wave * 6 + i) * 16;
#pragma unroll
        for (int r = 0; r < 8; ++r) qs[lh * 8 + r][n0 + ln] = acc[i][r];
    }
    __syncthreads();

    // ---- post: per (token,head,q|k) L2-norm * sqrt(scale) + RoPE ----
    {
        int u = tid >> 1, hs = tid & 1;
        int t = u >> 3, rem = u & 7, h = rem >> 1, sel = rem & 1;
        int m = m0 + t;
        int b = m >> 16, y = (m >> 8) & 255, x = m & 255;
        int col0 = sel * 256 + h * 64;
        float ss = 0.f;
        for (int d = 0; d < 64; ++d) { float v = qs[t][col0 + d]; ss += v * v; }
        float rn = sqrtf(attn_scale[h]) * rsqrtf(ss + 1e-6f);
        unsigned short* dst =
            (sel ? kn : qn) + ((size_t)(((b * 4 + h) * 256 + y) * 256 + x)) * 64;
        if (hs == 0) {
            size_t pidx = ((size_t)((b * 256 + y) * 256 + x)) * 2;
            float ph = pos[pidx + 0];
            float pw = pos[pidx + 1];
            const float LP = 1.1447298858494002f;       // ln(pi)
            const float LS = 2.302585092994046f / 32.f; // ln(10)/32
#pragma unroll
            for (int j = 0; j < 8; ++j) {
                float f = __expf(LP + LS * (float)(j * 4 + h));
                {   // d = j, theta = pos_h * f
                    float th = ph * f, c = __cosf(th), s = __sinf(th);
                    float x1 = qs[t][col0 + j] * rn, x2 = qs[t][col0 + 16 + j] * rn;
                    dst[j] = f2bf(x1 * c - x2 * s);
                    dst[16 + j] = f2bf(x2 * c + x1 * s);
                }
                {   // d = 8+j, theta = pos_w * f
                    float th = pw * f, c = __cosf(th), s = __sinf(th);
                    float x1 = qs[t][col0 + 8 + j] * rn, x2 = qs[t][col0 + 24 + j] * rn;
                    dst[8 + j] = f2bf(x1 * c - x2 * s);
                    dst[24 + j] = f2bf(x2 * c + x1 * s);
                }
            }
        } else {
            for (int d = 32; d < 64; ++d) dst[d] = f2bf(qs[t][col0 + d] * rn);
        }
    }
    // ---- post: V pass-through ----
    {
        int t = tid >> 4, seg = tid & 15;
        int m = m0 + t;
        int b = m >> 16, y = (m >> 8) & 255, x = m & 255;
        int fi = seg * 16;
        int h = fi >> 6, d0 = fi & 63;
        unsigned short tmp[16] __attribute__((aligned(16)));
#pragma unroll
        for (int e = 0; e < 16; ++e) tmp[e] = f2bf(qs[t][512 + fi + e]);
        unsigned short* dv =
            vv + ((size_t)(((b * 4 + h) * 256 + y) * 256 + x)) * 64 + d0;
        *(U4*)dv = *(U4*)tmp;
        *(U4*)(dv + 8) = *(U4*)(tmp + 8);
    }
}

// ---------- K4: windowed attention, one workgroup per (b,h,hw,ww) window ----------
__global__ __launch_bounds__(256) void k4_attn(const unsigned short* __restrict__ qn,
                                               const unsigned short* __restrict__ kn,
                                               const unsigned short* __restrict__ vv,
                                               unsigned short* __restrict__ o) {
    __shared__ unsigned short qsm[64][64];
    __shared__ unsigned short ksm[64][64];
    __shared__ unsigned short vts[64][64];   // transposed: vts[d][token]
    __shared__ float sc[64][65];
    __shared__ unsigned short ps[64][64];

    int tid = threadIdx.x;
    int wid = blockIdx.x;
    int ww = wid & 31, hw = (wid >> 5) & 31, h = (wid >> 10) & 3, b = wid >> 12;

    // ---- load window tiles (64 tokens x 64 dims), shift = 4 (roll +4 => src = idx-4) ----
    {
        int l = tid >> 2, part = tid & 3;
        int y = (hw * 8 + (l >> 3) + 252) & 255;
        int x = (ww * 8 + (l & 7) + 252) & 255;
        size_t base = ((size_t)(((b * 4 + h) * 256 + y) * 256 + x)) * 64 + part * 16;
        const U4* q4 = (const U4*)(qn + base);
        *(U4*)&qsm[l][part * 16] = q4[0];
        *(U4*)&qsm[l][part * 16 + 8] = q4[1];
        const U4* k4 = (const U4*)(kn + base);
        *(U4*)&ksm[l][part * 16] = k4[0];
        *(U4*)&ksm[l][part * 16 + 8] = k4[1];
        unsigned short tmp[16] __attribute__((aligned(16)));
        *(U4*)tmp = ((const U4*)(vv + base))[0];
        *(U4*)(tmp + 8) = ((const U4*)(vv + base))[1];
#pragma unroll
        for (int e = 0; e < 16; ++e) vts[part * 16 + e][l] = tmp[e];
    }
    __syncthreads();

    int wave = tid >> 5, lane = tid & 31, ln = lane & 15, lh = lane >> 4;
    bool top = (hw == 0), left = (ww == 0);

    // ---- S = Q @ K^T with shift mask ----
#pragma unroll
    for (int i2 = 0; i2 < 2; ++i2) {
        int tile = wave * 2 + i2, ti = tile >> 2, tj = tile & 3;
        v8f acc = zero8();
#pragma unroll
        for (int kk = 0; kk < 64; kk += 32) {
            FragBF a, bb;
            load_a_bf16(a, &qsm[ti * 16 + ln][0], kk + lh * 16);
            load_b_bf16(bb, &ksm[tj * 16 + ln][0], kk + lh * 16);
            acc = wmma_bf16(a, bb, acc);
        }
#pragma unroll
        for (int r = 0; r < 8; ++r) {
            int i = ti * 16 + lh * 8 + r, j = tj * 16 + ln;
            bool ok = (!left || (((i & 7) < 4) == ((j & 7) < 4))) &&
                      (!top || (((i >> 3) < 4) == ((j >> 3) < 4)));
            sc[i][j] = ok ? acc[r] : -1e9f;
        }
    }
    __syncthreads();

    // ---- row softmax ----
    if (tid < 64) {
        float mx = -3.4e38f;
        for (int j = 0; j < 64; ++j) mx = fmaxf(mx, sc[tid][j]);
        float s = 0.f;
        for (int j = 0; j < 64; ++j) { float e = __expf(sc[tid][j] - mx); sc[tid][j] = e; s += e; }
        float inv = 1.f / s;
        for (int j = 0; j < 64; ++j) ps[tid][j] = f2bf(sc[tid][j] * inv);
    }
    __syncthreads();

    // ---- O = P @ V, scatter to [b,y,x,D] (un-shift == source coords) ----
#pragma unroll
    for (int i2 = 0; i2 < 2; ++i2) {
        int tile = wave * 2 + i2, ti = tile >> 2, tj = tile & 3;
        v8f acc = zero8();
#pragma unroll
        for (int kk = 0; kk < 64; kk += 32) {
            FragBF a, bb;
            load_a_bf16(a, &ps[ti * 16 + ln][0], kk + lh * 16);
            load_b_bf16(bb, &vts[tj * 16 + ln][0], kk + lh * 16);
            acc = wmma_bf16(a, bb, acc);
        }
#pragma unroll
        for (int r = 0; r < 8; ++r) {
            int i = ti * 16 + lh * 8 + r, d = tj * 16 + ln;
            int y = (hw * 8 + (i >> 3) + 252) & 255;
            int x = (ww * 8 + (i & 7) + 252) & 255;
            o[((size_t)((b * 256 + y) * 256 + x)) * 256 + h * 64 + d] = f2bf(acc[r]);
        }
    }
}

// ---------- K5: out = skip + gate * (o @ out_w.T) ----------
__global__ __launch_bounds__(256) void k5_out(const unsigned short* __restrict__ o,
                                              const unsigned short* __restrict__ outw_bf,
                                              const float* __restrict__ x_in,
                                              const float* __restrict__ c_ws,
                                              float* __restrict__ out) {
    int tid = threadIdx.x;
    int wave = tid >> 5, lane = tid & 31, ln = lane & 15, lh = lane >> 4;
    int m0 = blockIdx.x * 16;
    int n0a = (wave * 2) * 16, n0b = (wave * 2 + 1) * 16;
    v8f acc0 = zero8(), acc1 = zero8();
    const unsigned short* ar = o + (size_t)(m0 + ln) * 256;
    for (int kk = 0; kk < 256; kk += 32) {
        FragBF a;
        load_a_bf16(a, ar, kk + lh * 16);
        FragBF b0, b1;
        load_b_bf16(b0, outw_bf + (size_t)(n0a + ln) * 256, kk + lh * 16);
        load_b_bf16(b1, outw_bf + (size_t)(n0b + ln) * 256, kk + lh * 16);
        acc0 = wmma_bf16(a, b0, acc0);
        acc1 = wmma_bf16(a, b1, acc1);
    }
    int bidx = m0 >> 16;
#pragma unroll
    for (int r = 0; r < 8; ++r) {
        int m = m0 + lh * 8 + r;
        {
            int n = n0a + ln;
            float g = c_ws[bidx * 768 + 512 + n];
            size_t off = (size_t)m * 256 + n;
            out[off] = x_in[off] + g * acc0[r];
        }
        {
            int n = n0b + ln;
            float g = c_ws[bidx * 768 + 512 + n];
            size_t off = (size_t)m * 256 + n;
            out[off] = x_in[off] + g * acc1[r];
        }
    }
}

// ---------- launch ----------
extern "C" void kernel_launch(void* const* d_in, const int* in_sizes, int n_in,
                              void* d_out, int out_size, void* d_ws, size_t ws_size,
                              hipStream_t stream) {
    const float* x_in       = (const float*)d_in[0];
    const float* pos        = (const float*)d_in[1];
    const float* cond       = (const float*)d_in[2];
    const float* norm_scale = (const float*)d_in[3];
    const float* mod_w      = (const float*)d_in[4];
    const float* mod_b      = (const float*)d_in[5];
    const float* qkv_w      = (const float*)d_in[6];
    const float* attn_scale = (const float*)d_in[7];
    const float* out_w      = (const float*)d_in[8];
    // d_in[9]=window_size(8), d_in[10]=window_shift(4): hardcoded.

    char* wsb = (char*)d_ws;
    const size_t SEG = (size_t)131072 * 256 * 2; // 64 MiB per bf16 tensor
    float* c_ws = (float*)wsb;
    unsigned short* xn_ws = (unsigned short*)(wsb + 8192);
    unsigned short* qn_ws = (unsigned short*)(wsb + 8192 + SEG);
    unsigned short* kn_ws = (unsigned short*)(wsb + 8192 + 2 * SEG);
    unsigned short* v_ws  = (unsigned short*)(wsb + 8192 + 3 * SEG);
    unsigned short* qkvw_bf = (unsigned short*)(wsb + 8192 + 4 * SEG);
    unsigned short* outw_bf = qkvw_bf + (size_t)768 * 256;
    unsigned short* o_ws  = xn_ws;  // xn dead after k3; reuse for attention output

    k0_cvt<<<192, 256, 0, stream>>>(qkv_w, qkvw_bf, 768 * 256);
    k0_cvt<<<64, 256, 0, stream>>>(out_w, outw_bf, 256 * 256);
    k1_mod<<<6, 256, 0, stream>>>(cond, mod_w, mod_b, c_ws);
    k2_xn<<<16384, 256, 0, stream>>>(x_in, norm_scale, c_ws, xn_ws);
    k3_qkv<<<8192, 256, 0, stream>>>(qkvw_bf, pos, attn_scale, xn_ws, qn_ws, kn_ws, v_ws);
    k4_attn<<<8192, 256, 0, stream>>>(qn_ws, kn_ws, v_ws, o_ws);
    k5_out<<<8192, 256, 0, stream>>>(o_ws, outw_bf, x_in, c_ws, (float*)d_out);
}
